// LSTM_52037823758737
// MI455X (gfx1250) — matrix operand order repaired and checked by reading, not compiled, not run on hardware
//
#include <hip/hip_runtime.h>
#include <hip/hip_bf16.h>

// ---------------------------------------------------------------------------
// LSTM on MI455X (gfx1250), bf16 WMMA path.
//   Kernel 1: Zx[t][b][n] = xs @ W_ih^T + bias   (time-parallel, 137 GFLOP)
//             64x64 block tile, 8 waves x (16x32) C-region, 2 wmma/stage/wave.
//   Kernel 2: persistent recurrence, 32 WGs, W_hh slice resident in LDS,
//             h broadcast via GLOBAL_LOAD_ASYNC_TO_LDS_B128 (ASYNCcnt path).
// Workspace: Zx fp32 (512 MB) + h ping-pong bf16 (128 KB) + 2 barrier ints.
// ---------------------------------------------------------------------------

typedef __attribute__((ext_vector_type(16))) __bf16 bf16x16;
typedef __attribute__((ext_vector_type(8)))  float  floatx8;

#define B_  64
#define T_  1024
#define D_  512
#define H_  512
#define NG  2048   // 4*H

// Pack two fp32 -> one dword of bf16 (compiler fuses to v_cvt_pk_bf16_f32).
__device__ inline unsigned pk2(float a, float b) {
  union { __bf16 h[2]; unsigned u; } p;
  p.h[0] = (__bf16)a; p.h[1] = (__bf16)b;
  return p.u;
}

// ---- WMMA fragment loaders (ISA 7.12.2 layouts, wave32) -------------------
// A (16x32 bf16): lane L -> m = L&15 ; VGPR v<4: K = 2v + 8*(L>>4),
//                 VGPR v>=4: K = 16 + 2(v-4) + 8*(L>>4).  Row-major [m][k] LDS.
__device__ inline bf16x16 load_a_frag(const __bf16* A, int ldk) {
  const int lane = threadIdx.x & 31;
  const int kh = (lane >> 4) * 8;
  const __bf16* row = A + (lane & 15) * ldk;
  union { bf16x16 v; unsigned u[8]; } r;
#pragma unroll
  for (int v = 0; v < 4; ++v)
    r.u[v] = *(const unsigned*)(row + kh + 2 * v);
#pragma unroll
  for (int v = 0; v < 4; ++v)
    r.u[4 + v] = *(const unsigned*)(row + 16 + kh + 2 * v);
  return r.v;
}

// B (32x16 bf16, stored as W row-major [n][k]): lane L -> n = L&15,
// VGPR v<4: K = 2v + 8*(L>>4), VGPR v>=4: K = 16 + 2(v-4) + 8*(L>>4).
__device__ inline bf16x16 load_b_frag(const __bf16* Bn, int ldk) {
  const int lane = threadIdx.x & 31;
  const int kb = (lane >> 4) * 8;
  const __bf16* row = Bn + (lane & 15) * ldk;
  union { bf16x16 v; unsigned u[8]; } r;
#pragma unroll
  for (int v = 0; v < 8; ++v)
    r.u[v] = *(const unsigned*)(row + kb + 2 * v + (v >= 4 ? 8 : 0));
  return r.v;
}

__device__ inline float sigf(float x) { return 1.0f / (1.0f + __expf(-x)); }

// ---- Kernel 1: Zx = xs @ W_ih^T + bias ------------------------------------
// Block 256 thr (8 waves). Block tile 64(M) x 64(N); wave w: mt=w&3, nh=w>>2,
// wave owns 16x32 (two 16x16 C tiles, A-frag reused). 2 wmma per K-stage.
__global__ __launch_bounds__(256)
void lstm_xgemm(const float* __restrict__ xs, const float* __restrict__ Wih,
                const float* __restrict__ bias, float* __restrict__ Zx) {
  const int m0 = blockIdx.y * 64;   // M over B*T (m = b*T + t)
  const int n0 = blockIdx.x * 64;   // N over 4H
  constexpr int LK = 36;            // 32 + 4 pad bf16 (8B-aligned rows)
  __shared__ __bf16 As[64 * LK];
  __shared__ __bf16 Bs[64 * LK];

  const int tid  = threadIdx.x;
  const int wave = tid >> 5;
  const int mt = wave & 3, nh = wave >> 2;

  floatx8 acc0 = {}, acc1 = {};
  for (int k0 = 0; k0 < D_; k0 += 32) {
    // Stage A: 64 rows x 32 k; float4 load -> packed bf16 b64 store.
#pragma unroll
    for (int i = 0; i < 2; ++i) {
      int e = tid + i * 256;              // 0..511 : row = e/8, 4-float col
      int m = e >> 3, c4 = (e & 7) * 4;
      float4 f = *(const float4*)(xs + (size_t)(m0 + m) * D_ + k0 + c4);
      uint2 p; p.x = pk2(f.x, f.y); p.y = pk2(f.z, f.w);
      *(uint2*)(As + m * LK + c4) = p;
    }
    // Stage B: W_ih rows n0..n0+63 (n-major, k-contiguous).
#pragma unroll
    for (int i = 0; i < 2; ++i) {
      int e = tid + i * 256;
      int n = e >> 3, c4 = (e & 7) * 4;
      float4 f = *(const float4*)(Wih + (size_t)(n0 + n) * D_ + k0 + c4);
      uint2 p; p.x = pk2(f.x, f.y); p.y = pk2(f.z, f.w);
      *(uint2*)(Bs + n * LK + c4) = p;
    }
    __syncthreads();
    bf16x16 a  = load_a_frag(As + mt * 16 * LK, LK);
    bf16x16 b0 = load_b_frag(Bs + (nh * 32 +  0) * LK, LK);
    bf16x16 b1 = load_b_frag(Bs + (nh * 32 + 16) * LK, LK);
    acc0 = __builtin_amdgcn_wmma_f32_16x16x32_bf16(false, a, false, b0,
                                                   (short)0, acc0, false, false);
    acc1 = __builtin_amdgcn_wmma_f32_16x16x32_bf16(false, a, false, b1,
                                                   (short)0, acc1, false, false);
    __syncthreads();
  }
  // Epilogue: permute M=(b*T+t) -> store Zx[t][b][n], fold bias.
  const int lane = tid & 31;
  const int ln = lane & 15, lh = lane >> 4;
#pragma unroll
  for (int tn = 0; tn < 2; ++tn) {
    const int n = n0 + nh * 32 + tn * 16 + ln;
    const float bn = bias[n];
    const floatx8 a = tn ? acc1 : acc0;
#pragma unroll
    for (int r = 0; r < 8; ++r) {
      int m = m0 + mt * 16 + r + 8 * lh;
      int bb = m >> 10;                 // / T_
      int tt = m & (T_ - 1);
      Zx[((size_t)tt * B_ + bb) * NG + n] = a[r] + bn;
    }
  }
}

// ---- grid-wide barrier (persistent kernel, all 32 WGs resident) ------------
__device__ inline void gridbar(int* __restrict__ bar, int nblk) {
  __syncthreads();
  if (threadIdx.x == 0) {
    int gen  = __hip_atomic_load(&bar[1], __ATOMIC_ACQUIRE, __HIP_MEMORY_SCOPE_AGENT);
    int prev = __hip_atomic_fetch_add(&bar[0], 1, __ATOMIC_ACQ_REL, __HIP_MEMORY_SCOPE_AGENT);
    if (prev == nblk - 1) {
      __hip_atomic_store(&bar[0], 0, __ATOMIC_RELAXED, __HIP_MEMORY_SCOPE_AGENT);
      __hip_atomic_fetch_add(&bar[1], 1, __ATOMIC_ACQ_REL, __HIP_MEMORY_SCOPE_AGENT);
    } else {
      while (__hip_atomic_load(&bar[1], __ATOMIC_ACQUIRE, __HIP_MEMORY_SCOPE_AGENT) == gen)
        __builtin_amdgcn_s_sleep(2);
    }
  }
  __syncthreads();
}

// ---- Kernel 2: persistent recurrence --------------------------------------
// 32 blocks x 512 thr (16 waves). Block owns H-columns [j0, j0+16) across all
// 4 gates (64 N-cols). Wave w: mt=w&3 (batch tile), gate=w>>2. W_hh slice
// stays in LDS all 1024 steps. z accumulates Zx via C-frag init.
// h broadcast: raw bf16 bytes, global -> LDS on the async DMA path.
#define LDK 520   // 512 + 8 pad bf16: rows 1040 B (16B multiple, bank-clean)

__global__ __launch_bounds__(512)
void lstm_recur(const float* __restrict__ Zx, const float* __restrict__ Whh,
                float* __restrict__ out, __bf16* __restrict__ hglob,
                int* __restrict__ bar) {
  extern __shared__ char smem[];
  __bf16* Ws   = (__bf16*)smem;                 // [64][LDK]  W_hh slice (65 KB)
  __bf16* hs   = Ws + 64 * LDK;                 // [64][LDK]  h staging  (65 KB)
  float*  zbuf = (float*)(hs + 64 * LDK);       // [4][64][16] gate preacts (16 KB)

  const int tid = threadIdx.x;
  const int j0  = blockIdx.x * 16;

  // Stage W_hh slice once: Ws[g*16+jl][k] = bf16(W_hh[g*512 + j0 + jl][k]).
  for (int e = tid; e < 64 * (H_ / 4); e += 512) {
    int nl = e >> 7, c4 = (e & 127) * 4;
    int g = nl >> 4, jl = nl & 15;
    float4 f = *(const float4*)(Whh + (size_t)(g * H_ + j0 + jl) * H_ + c4);
    uint2 p; p.x = pk2(f.x, f.y); p.y = pk2(f.z, f.w);
    *(uint2*)(Ws + nl * LDK + c4) = p;
  }
  __syncthreads();

  const int wave = tid >> 5;
  const int mt   = wave & 3;
  const int gate = wave >> 2;
  const int lane = tid & 31;
  const int ln   = lane & 15;
  const int lh   = lane >> 4;
  const int ng   = gate * H_ + j0 + ln;         // global gate column

  float creg[2] = {0.0f, 0.0f};                 // cell state, 2 (b,j) per thread

  for (int t = 0; t < T_; ++t) {
    if (t > 0) {
      // Async DMA broadcast of h (bf16, 64 KB): 4096 16B chunks / 512 thr.
      const char* hsrc = (const char*)(hglob + (size_t)((t - 1) & 1) * (B_ * H_));
      char*       hdst = (char*)hs;
#pragma unroll
      for (int i = 0; i < 8; ++i) {
        int c   = tid + i * 512;                // chunk id
        int row = c >> 6, col = c & 63;         // 64 chunks per 1024B row
        unsigned    lofs = (unsigned)(uintptr_t)(hdst + row * (LDK * 2) + col * 16);
        const void* gsrc = hsrc + row * 1024 + col * 16;
        asm volatile("global_load_async_to_lds_b128 %0, %1, off"
                     :: "v"(lofs), "v"(gsrc) : "memory");
      }
      asm volatile("s_wait_asynccnt 0x0" ::: "memory");
    }
    __syncthreads();

    // C-frag initialized from precomputed Zx (folds x@W_ih^T + b for free).
    const float* zrow = Zx + (size_t)t * B_ * NG;
    floatx8 acc;
#pragma unroll
    for (int r = 0; r < 8; ++r)
      acc[r] = zrow[(size_t)(mt * 16 + r + 8 * lh) * NG + ng];

    if (t > 0) {
      const __bf16* arow = hs + (mt * 16) * LDK;
      const __bf16* brow = Ws + (gate * 16) * LDK;
#pragma unroll
      for (int kk = 0; kk < H_; kk += 32) {
        bf16x16 a = load_a_frag(arow + kk, LDK);
        bf16x16 b = load_b_frag(brow + kk, LDK);
        acc = __builtin_amdgcn_wmma_f32_16x16x32_bf16(false, a, false, b,
                                                      (short)0, acc, false, false);
      }
    }
#pragma unroll
    for (int r = 0; r < 8; ++r)                 // z tile -> LDS for gate fusion
      zbuf[(gate * 64 + (mt * 16 + r + 8 * lh)) * 16 + ln] = acc[r];
    __syncthreads();

    // Elementwise gates: i,f,g,o live in 4 different waves -> via zbuf.
    __bf16* hout = hglob + (size_t)(t & 1) * (B_ * H_);
#pragma unroll
    for (int p = 0; p < 2; ++p) {
      int idx = tid + p * 512;                  // (b, jl) pair
      int b = idx >> 4, jl = idx & 15;
      float zi = zbuf[(0 * 64 + b) * 16 + jl];
      float zf = zbuf[(1 * 64 + b) * 16 + jl];
      float zg = zbuf[(2 * 64 + b) * 16 + jl];
      float zo = zbuf[(3 * 64 + b) * 16 + jl];
      float c = sigf(zf) * creg[p] + sigf(zi) * tanhf(zg);
      float h = sigf(zo) * tanhf(c);
      creg[p] = c;
      hout[b * H_ + j0 + jl] = (__bf16)h;
      if (t == T_ - 1) {
        out[b * H_ + j0 + jl]           = h;    // h output
        out[B_ * H_ + b * H_ + j0 + jl] = c;    // c output
      }
    }
    __threadfence();
    gridbar(bar, gridDim.x);
  }
}

__global__ void lstm_bar_init(int* bar) { bar[0] = 0; bar[1] = 0; }

extern "C" void kernel_launch(void* const* d_in, const int* in_sizes, int n_in,
                              void* d_out, int out_size, void* d_ws, size_t ws_size,
                              hipStream_t stream) {
  const float* xs   = (const float*)d_in[0];   // [B,T,D]
  const float* Wih  = (const float*)d_in[1];   // [4H,D]
  const float* Whh  = (const float*)d_in[2];   // [4H,H]
  const float* bias = (const float*)d_in[3];   // [4H]
  float* out = (float*)d_out;                  // h [B,H] then c [B,H]

  char* ws = (char*)d_ws;
  const size_t zx_bytes = (size_t)T_ * B_ * NG * sizeof(float);   // 512 MB
  float*  Zx    = (float*)ws;
  __bf16* hglob = (__bf16*)(ws + zx_bytes);                       // 2*B*H bf16
  int*    bar   = (int*)(ws + zx_bytes + (size_t)2 * B_ * H_ * sizeof(__bf16));

  lstm_bar_init<<<1, 1, 0, stream>>>(bar);

  dim3 g1(NG / 64, (B_ * T_) / 64);
  lstm_xgemm<<<g1, 256, 0, stream>>>(xs, Wih, bias, Zx);

  const size_t shmem = (size_t)2 * 64 * LDK * sizeof(__bf16)      // Ws + hs
                     + (size_t)4 * 64 * 16 * sizeof(float);       // zbuf
  lstm_recur<<<32, 512, shmem, stream>>>(Zx, Whh, out, hglob, bar);
}
